// Matcher_29222957482861
// MI455X (gfx1250) — matrix operand order repaired and checked by reading, not compiled, not run on hardware
//
#include <hip/hip_runtime.h>

#define KN1 8192
#define KN2 8192
#define KD  128
#define TILE 128
#define LDSP 136   // padded LDS row stride in halves (272 B: 16B-aligned, bank-conflict-free)

typedef __attribute__((ext_vector_type(16))) _Float16 v16h;
typedef __attribute__((ext_vector_type(8)))  _Float16 v8h;
typedef __attribute__((ext_vector_type(8)))  float    v8f;

struct alignas(8) H4 { _Float16 v[4]; };

// Monotonic float -> u32 key (total order matching float compare)
static __device__ __forceinline__ unsigned fkey(float f) {
  unsigned u = __float_as_uint(f);
  return u ^ ((u & 0x80000000u) ? 0xFFFFFFFFu : 0x80000000u);
}
// Pack (score, index) so u64-max == argmax with first-index tie-breaking
static __device__ __forceinline__ unsigned long long packmax(float f, int idx) {
  return ((unsigned long long)fkey(f) << 32) | (unsigned)(~idx);
}
static __device__ __forceinline__ unsigned long long shflx64(unsigned long long v, int m) {
  unsigned lo = __shfl_xor((unsigned)v, m, 32);
  unsigned hi = __shfl_xor((unsigned)(v >> 32), m, 32);
  return ((unsigned long long)hi << 32) | lo;
}
static __device__ __forceinline__ v16h join16(v8h a, v8h b) {
  union { v16h w; v8h h[2]; } u;
  u.h[0] = a; u.h[1] = b;
  return u.w;
}
// fp32 -> f16 hi/lo split: f ~= hi + lo with ~24 mantissa bits total
static __device__ __forceinline__ void split4(const float4 f, H4& hi, H4& lo) {
  const float x[4] = {f.x, f.y, f.z, f.w};
#pragma unroll
  for (int j = 0; j < 4; ++j) {
    _Float16 h = (_Float16)x[j];
    hi.v[j] = h;
    lo.v[j] = (_Float16)(x[j] - (float)h);
  }
}

__global__ void init_kernel(unsigned long long* best) {
  int i = blockIdx.x * 256 + threadIdx.x;
  if (i < KN1 + KN2) best[i] = 0ull;
}

// One-shot fp32 -> (hi,lo) f16 plane conversion for both descriptor matrices.
__global__ __launch_bounds__(256)
void convert_kernel(const float* __restrict__ d1, const float* __restrict__ d2,
                    _Float16* __restrict__ g1h, _Float16* __restrict__ g1l,
                    _Float16* __restrict__ g2h, _Float16* __restrict__ g2l) {
  int idx = blockIdx.x * 256 + threadIdx.x;     // float4 id over both matrices
  const int per = KN1 * (KD / 4);               // 262144 float4 per matrix
  int m = idx >= per;
  int i = idx - m * per;
  const float* src = m ? d2 : d1;
  _Float16* dh = m ? g2h : g1h;
  _Float16* dl = m ? g2l : g1l;
  float4 v = ((const float4*)src)[i];
  H4 h, l;
  split4(v, h, l);
  *(H4*)(dh + (size_t)i * 4) = h;
  *(H4*)(dl + (size_t)i * 4) = l;
}

// Shared compute + fused two-sided argmax (LDS planes already filled, barrier done).
static __device__ __forceinline__ void tile_compute_reduce(
    const _Float16* Ah, const _Float16* Al,
    const _Float16* Bh, const _Float16* Bl,
    unsigned long long* colred, int rowBase, int colBase,
    unsigned long long* __restrict__ rowBest, unsigned long long* __restrict__ colBest) {
  const int tid  = threadIdx.x;
  const int wave = tid >> 5;
  const int lane = tid & 31;
  const int hn   = lane & 15;
  const int hh   = lane >> 4;

  v8f acc[8];
  const v8f vzero = {0.f, 0.f, 0.f, 0.f, 0.f, 0.f, 0.f, 0.f};
#pragma unroll
  for (int t = 0; t < 8; ++t) acc[t] = vzero;

  const int arow = wave * 16 + hn;
#pragma unroll
  for (int ks = 0; ks < 4; ++ks) {
    const _Float16* ab = Ah + arow * LDSP + ks * 32;
    const _Float16* al = Al + arow * LDSP + ks * 32;
    v16h aH = join16(*(const v8h*)(ab + hh * 16), *(const v8h*)(ab + hh * 16 + 8));
    v16h aL = join16(*(const v8h*)(al + hh * 16), *(const v8h*)(al + hh * 16 + 8));
#pragma unroll
    for (int t = 0; t < 8; ++t) {
      const _Float16* bb = Bh + (t * 16 + hn) * LDSP + ks * 32 + hh * 16;
      const _Float16* bl = Bl + (t * 16 + hn) * LDSP + ks * 32 + hh * 16;
      v16h bH = join16(*(const v8h*)bb, *(const v8h*)(bb + 8));
      v16h bL = join16(*(const v8h*)bl, *(const v8h*)(bl + 8));
      acc[t] = __builtin_amdgcn_wmma_f32_16x16x32_f16(false, aH, false, bH, (short)0, acc[t], false, false);
      acc[t] = __builtin_amdgcn_wmma_f32_16x16x32_f16(false, aH, false, bL, (short)0, acc[t], false, false);
      acc[t] = __builtin_amdgcn_wmma_f32_16x16x32_f16(false, aL, false, bH, (short)0, acc[t], false, false);
    }
  }

  // Row argmax: acc[t][i] is sim[row = i + 8*hh, col = t*16 + hn]
#pragma unroll
  for (int i = 0; i < 8; ++i) {
    unsigned long long best = 0ull;
#pragma unroll
    for (int t = 0; t < 8; ++t) {
      unsigned long long p = packmax(acc[t][i], colBase + t * 16 + hn);
      if (p > best) best = p;
    }
#pragma unroll
    for (int m = 1; m < 16; m <<= 1) {
      unsigned long long o = shflx64(best, m);
      if (o > best) best = o;
    }
    if (hn == 0) {
      int row = rowBase + wave * 16 + i + 8 * hh;
      atomicMax(&rowBest[row], best);
    }
  }

  // Column argmax over this wave's 16 rows
#pragma unroll
  for (int t = 0; t < 8; ++t) {
    unsigned long long best = 0ull;
#pragma unroll
    for (int i = 0; i < 8; ++i) {
      unsigned long long p = packmax(acc[t][i], rowBase + wave * 16 + i + 8 * hh);
      if (p > best) best = p;
    }
    unsigned long long o = shflx64(best, 16);
    if (o > best) best = o;
    if (hh == 0) colred[wave * 128 + t * 16 + hn] = best;
  }
  __syncthreads();
  if (tid < 128) {
    unsigned long long best = 0ull;
#pragma unroll
    for (int w = 0; w < 8; ++w) {
      unsigned long long p = colred[w * 128 + tid];
      if (p > best) best = p;
    }
    atomicMax(&colBest[colBase + tid], best);
  }
}

// Fast path: planes pre-converted in global; fill LDS via async global->LDS DMA.
__global__ __launch_bounds__(256)
void sim_tile_kernel_async(const _Float16* __restrict__ g1h, const _Float16* __restrict__ g1l,
                           const _Float16* __restrict__ g2h, const _Float16* __restrict__ g2l,
                           unsigned long long* __restrict__ rowBest,
                           unsigned long long* __restrict__ colBest) {
  __shared__ alignas(16) _Float16 Ah[TILE * LDSP];
  __shared__ alignas(16) _Float16 Al[TILE * LDSP];
  __shared__ alignas(16) _Float16 Bh[TILE * LDSP];
  __shared__ alignas(16) _Float16 Bl[TILE * LDSP];
  __shared__ unsigned long long colred[8 * 128];

  const int tid = threadIdx.x;
  const int rowBase = blockIdx.y * TILE;
  const int colBase = blockIdx.x * TILE;

  const _Float16* gAh = g1h + (size_t)rowBase * KD;
  const _Float16* gAl = g1l + (size_t)rowBase * KD;
  const _Float16* gBh = g2h + (size_t)colBase * KD;
  const _Float16* gBl = g2l + (size_t)colBase * KD;
  unsigned ldsAh = (unsigned)(size_t)Ah;
  unsigned ldsAl = (unsigned)(size_t)Al;
  unsigned ldsBh = (unsigned)(size_t)Bh;
  unsigned ldsBl = (unsigned)(size_t)Bl;

  // 128 rows x 16 chunks of 16B per plane; 8 chunks per thread per plane.
#pragma unroll
  for (int it = 0; it < 8; ++it) {
    int idx = tid + it * 256;
    int r = idx >> 4;
    int c = (idx & 15) * 8;
    unsigned lo = (unsigned)((r * LDSP + c) * 2);   // LDS byte offset (padded)
    unsigned go = (unsigned)((r * KD + c) * 2);     // global byte offset (dense)
    asm volatile("global_load_async_to_lds_b128 %0, %1, %2"
                 :: "v"(ldsAh + lo), "v"(go), "s"(gAh) : "memory");
    asm volatile("global_load_async_to_lds_b128 %0, %1, %2"
                 :: "v"(ldsAl + lo), "v"(go), "s"(gAl) : "memory");
    asm volatile("global_load_async_to_lds_b128 %0, %1, %2"
                 :: "v"(ldsBh + lo), "v"(go), "s"(gBh) : "memory");
    asm volatile("global_load_async_to_lds_b128 %0, %1, %2"
                 :: "v"(ldsBl + lo), "v"(go), "s"(gBl) : "memory");
  }
  asm volatile("s_wait_asynccnt 0x0" ::: "memory");
  __syncthreads();

  tile_compute_reduce(Ah, Al, Bh, Bl, colred, rowBase, colBase, rowBest, colBest);
}

// Fallback path: convert fp32 -> hi/lo in-kernel (used when d_ws is too small).
__global__ __launch_bounds__(256)
void sim_tile_kernel_conv(const float* __restrict__ d1, const float* __restrict__ d2,
                          unsigned long long* __restrict__ rowBest,
                          unsigned long long* __restrict__ colBest) {
  __shared__ alignas(16) _Float16 Ah[TILE * LDSP];
  __shared__ alignas(16) _Float16 Al[TILE * LDSP];
  __shared__ alignas(16) _Float16 Bh[TILE * LDSP];
  __shared__ alignas(16) _Float16 Bl[TILE * LDSP];
  __shared__ unsigned long long colred[8 * 128];

  const int tid = threadIdx.x;
  const int rowBase = blockIdx.y * TILE;
  const int colBase = blockIdx.x * TILE;

#pragma unroll
  for (int it = 0; it < 16; ++it) {
    int idx = tid + it * 256;
    int r = idx >> 5;
    int q = idx & 31;
    float4 a = ((const float4*)(d1 + (size_t)(rowBase + r) * KD))[q];
    float4 b = ((const float4*)(d2 + (size_t)(colBase + r) * KD))[q];
    H4 h, l;
    split4(a, h, l);
    *(H4*)(Ah + r * LDSP + q * 4) = h;
    *(H4*)(Al + r * LDSP + q * 4) = l;
    split4(b, h, l);
    *(H4*)(Bh + r * LDSP + q * 4) = h;
    *(H4*)(Bl + r * LDSP + q * 4) = l;
  }
  __syncthreads();

  tile_compute_reduce(Ah, Al, Bh, Bl, colred, rowBase, colBase, rowBest, colBest);
}

__global__ void finalize_kernel(const unsigned long long* __restrict__ rowBest,
                                const unsigned long long* __restrict__ colBest,
                                int* __restrict__ outM, float* __restrict__ outS) {
  int r = blockIdx.x * 256 + threadIdx.x;
  if (r >= KN1) return;
  unsigned long long p = rowBest[r];
  int col = (int)~(unsigned)p;
  unsigned key  = (unsigned)(p >> 32);
  unsigned bits = (key & 0x80000000u) ? (key ^ 0x80000000u) : ~key;
  unsigned long long q = colBest[col];
  int brow = (int)~(unsigned)q;
  outM[r] = (brow == r) ? col : -1;
  outS[r] = __uint_as_float(bits);
}

extern "C" void kernel_launch(void* const* d_in, const int* in_sizes, int n_in,
                              void* d_out, int out_size, void* d_ws, size_t ws_size,
                              hipStream_t stream) {
  (void)in_sizes; (void)n_in; (void)out_size;
  const float* d1 = (const float*)d_in[0];
  const float* d2 = (const float*)d_in[1];

  unsigned long long* rowBest = (unsigned long long*)d_ws;       // u64[KN1]
  unsigned long long* colBest = rowBest + KN1;                   // u64[KN2]
  const size_t bestBytes  = (size_t)(KN1 + KN2) * 8;             // 128 KB
  const size_t planeElems = (size_t)KN1 * KD;                    // 1M halves / plane
  const size_t needBytes  = bestBytes + 4 * planeElems * 2;      // ~8.1 MB

  init_kernel<<<(KN1 + KN2 + 255) / 256, 256, 0, stream>>>(rowBest);

  dim3 grid(KN2 / TILE, KN1 / TILE);                             // 64 x 64 tiles
  if (ws_size >= needBytes) {
    _Float16* g1h = (_Float16*)((char*)d_ws + bestBytes);
    _Float16* g1l = g1h + planeElems;
    _Float16* g2h = g1l + planeElems;
    _Float16* g2l = g2h + planeElems;
    convert_kernel<<<(2 * KN1 * (KD / 4)) / 256, 256, 0, stream>>>(d1, d2, g1h, g1l, g2h, g2l);
    sim_tile_kernel_async<<<grid, dim3(256), 0, stream>>>(g1h, g1l, g2h, g2l, rowBest, colBest);
  } else {
    sim_tile_kernel_conv<<<grid, dim3(256), 0, stream>>>(d1, d2, rowBest, colBest);
  }

  finalize_kernel<<<KN1 / 256, 256, 0, stream>>>(rowBest, colBest,
                                                 (int*)d_out, (float*)d_out + KN1);
}
